// TemporalGraphAttention_31018253811745
// MI455X (gfx1250) — matrix operand order repaired and checked by reading, not compiled, not run on hardware
//
#include <hip/hip_runtime.h>
#include <hip/hip_bf16.h>

// MI455X / gfx1250, wave32. WMMA f16 16x16x32 w/ f32 accum everywhere;
// weight tiles staged into LDS by the Tensor Data Mover (TDM).
//
// scores = (Q + T) K^T  (algebraic fusion of QK^T + TK^T)
// out = softmax(scores/8) V @ wo^T + bo ; attn matrix materialized fp32.

typedef __attribute__((ext_vector_type(16))) _Float16 v16h;
typedef __attribute__((ext_vector_type(8)))  _Float16 v8h;
typedef __attribute__((ext_vector_type(8)))  float    v8f;
typedef __attribute__((ext_vector_type(4)))  unsigned int v4u;
typedef __attribute__((ext_vector_type(8)))  int      v8i;
typedef __attribute__((ext_vector_type(4)))  int      v4i;

#define D_MODEL 1024
#define N_HEADS 16
#define D_HEAD  64
#define BATCH   2
#define SEQ     2048
#define M_TOT   (BATCH * SEQ)   // 4096 rows (b*s flattened)

#if __has_builtin(__builtin_amdgcn_tensor_load_to_lds)
#define USE_TDM 1
#endif

static __device__ __forceinline__ v8f wmma_f16(v16h a, v16h b, v8f c) {
  return __builtin_amdgcn_wmma_f32_16x16x32_f16(
      /*neg_a=*/false, a, /*neg_b=*/false, b,
      /*c_mod=*/(short)0, c, /*reuse_a=*/false, /*reuse_b=*/false);
}

static __device__ __forceinline__ v16h pack16(v8h lo, v8h hi) {
  return __builtin_shufflevector(lo, hi, 0, 1, 2, 3, 4, 5, 6, 7,
                                         8, 9, 10, 11, 12, 13, 14, 15);
}

// A fragment: 16(M) x 32(K), row-major mem[row*ld + k].
// lanes 0-15 (row=lane) hold K {0..7,16..23}; lanes 16-31 hold K {8..15,24..31}.
static __device__ __forceinline__ v16h load_a_g(const _Float16* base, int ld, int lane) {
  int r  = lane & 15;
  int hi = (lane >> 4) << 3;
  const _Float16* p = base + (size_t)r * ld + hi;
  v8h lo = *(const v8h*)p;
  v8h hh = *(const v8h*)(p + 16);
  return pack16(lo, hh);
}

// B fragment: 32(K) x 16(N); memory holds N rows each K-contiguous: mem[n*ld + k].
// lanes 0-15 (N=lane) hold K {0..7,16..23}; lanes 16-31 hold K {8..15,24..31}.
static __device__ __forceinline__ v16h load_b_g(const _Float16* base, int ld, int lane) {
  int n  = lane & 15;
  int k0 = (lane >> 4) << 3;
  const _Float16* p = base + (size_t)n * ld + k0;
  return pack16(*(const v8h*)p, *(const v8h*)(p + 16));
}

// ---------------------------------------------------------------- TDM staging
// Stage a rows x cols (f16) tile, row stride row_stride (elems), into LDS.
static __device__ __forceinline__ void stage_tile(_Float16* dst, const _Float16* src,
                                                  int rows, int cols, int row_stride,
                                                  int tid, int wid) {
#ifdef USE_TDM
  if (wid == 0) {  // TDM issues once per wave; scalar branch on wave id
    unsigned lds = (unsigned)(unsigned long long)(size_t)dst;   // low 32 = LDS byte addr
    unsigned long long ga = (unsigned long long)(size_t)src;
    // D# group 0: count=1 | lds_addr | global_addr[56:0] | type=2
    v4u g0;
    g0.x = 1u;
    g0.y = lds;
    g0.z = (unsigned)(ga & 0xffffffffull);
    g0.w = (unsigned)((ga >> 32) & 0x1ffffffull) | (2u << 30);
    // D# group 1: data_size=2B; tensor dims = tile dims; dim0 stride = row_stride
    unsigned td0 = (unsigned)cols, td1 = (unsigned)rows;
    unsigned long long st0 = (unsigned long long)row_stride;
    v8i g1;
    g1[0] = 0x00010000;                                         // data_size=1 (2 bytes)
    g1[1] = (int)((td0 & 0xffffu) << 16);                       // tensor_dim0[15:0]
    g1[2] = (int)(((td0 >> 16) & 0xffffu) | ((td1 & 0xffffu) << 16));
    g1[3] = (int)(((td1 >> 16) & 0xffffu) | ((td0 & 0xffffu) << 16)); // tile_dim0=cols
    g1[4] = (int)(td1 & 0xffffu);                               // tile_dim1=rows
    g1[5] = (int)(st0 & 0xffffffffull);                         // dim0_stride lo
    g1[6] = (int)((st0 >> 32) & 0xffffull);                     // dim0_stride hi
    g1[7] = 0;
    v4i g2 = {0, 0, 0, 0}, g3 = {0, 0, 0, 0};
    v8i g4 = {0, 0, 0, 0, 0, 0, 0, 0};
    // amdgpu-toolchain (clang-23) 6-arg form:
    // (uint32x4 g0, int32x8 g1, int32x4 g2, int32x4 g3, int32x8 g4, i32 cpol)
    __builtin_amdgcn_tensor_load_to_lds(g0, g1, g2, g3, g4, 0);
  }
  (void)tid;
#else
  int vpr = cols >> 3;   // v8h per row
  for (int i = tid; i < rows * vpr; i += 256) {
    int r = i / vpr, cc = i - r * vpr;
    *(v8h*)(dst + (size_t)r * cols + (size_t)cc * 8) =
        *(const v8h*)(src + (size_t)r * row_stride + (size_t)cc * 8);
  }
  (void)wid;
#endif
}

static __device__ __forceinline__ void tile_wait(int wid) {
#ifdef USE_TDM
  if (wid == 0) __builtin_amdgcn_s_wait_tensorcnt(0);
#else
  (void)wid;
#endif
}

// ---------------------------------------------------------------- prep kernels
__global__ void cvt_f32_to_f16(const float* __restrict__ src,
                               _Float16* __restrict__ dst, int n) {
  int i = blockIdx.x * blockDim.x + threadIdx.x;
  if (i < n) dst[i] = (_Float16)src[i];
}

__global__ void prep_kernel(const float* __restrict__ x, const float* __restrict__ ts,
                            const float* __restrict__ tw, const float* __restrict__ tb,
                            _Float16* __restrict__ x16, _Float16* __restrict__ te16,
                            _Float16* __restrict__ xk16, int n) {
  int i = blockIdx.x * blockDim.x + threadIdx.x;
  if (i >= n) return;
  int d  = i & (D_MODEL - 1);
  int bs = i >> 10;
  float te = cosf(ts[bs] * tw[d] + tb[d]);
  float xv = x[i];
  x16[i]  = (_Float16)xv;
  te16[i] = (_Float16)te;
  xk16[i] = (_Float16)(xv + te);
}

// ---------------------------------------------------------------- WMMA GEMM
// C[m,n] = A[m,:] . W[n,:] (+ A2.W2 if DUAL) + bias(n)
// Block = 128(M) x 64(N); 8 waves, each 16(M) x 64(N) = 4 accumulators.
// Weight tile (64 x 128 f16, 16 KB) staged to LDS by TDM each K-chunk.
// MODE 0: f16 [M,N]; MODE 1: f16 transposed per-batch [B, N, SEQ]; MODE 2: f32 [M,N].
template <int DUAL, int MODE>
__global__ __launch_bounds__(256) void gemm_wmma(
    const _Float16* __restrict__ A, const _Float16* __restrict__ W,
    const _Float16* __restrict__ A2, const _Float16* __restrict__ W2,
    const float* __restrict__ bias, const float* __restrict__ bias2,
    void* __restrict__ outv, int M, int N, int K) {
  __shared__ __align__(16) _Float16 sW[2][64 * 128];   // 32 KB

  int tid  = threadIdx.x;
  int lane = tid & 31;
  int wid  = tid >> 5;                 // 8 waves -> 8 M sub-tiles
  int ntb  = N >> 6;
  int bm   = blockIdx.x / ntb;
  int bn   = blockIdx.x % ntb;
  int m0   = bm * 128 + wid * 16;
  int n0   = bn * 64;

  v8f cacc[4] = {{}, {}, {}, {}};

  for (int kc = 0; kc < K; kc += 128) {
    __syncthreads();                   // previous chunk fully consumed
    stage_tile(&sW[0][0], W + (size_t)n0 * K + kc, 64, 128, K, tid, wid);
    if (DUAL) stage_tile(&sW[1][0], W2 + (size_t)n0 * K + kc, 64, 128, K, tid, wid);
    tile_wait(wid);
    __syncthreads();                   // LDS tile visible to all waves

    for (int ks = 0; ks < 128; ks += 32) {
      v16h a = load_a_g(A + (size_t)m0 * K + kc + ks, K, lane);
      __builtin_prefetch(A + (size_t)m0 * K + kc + ks + 128, 0, 1);
#pragma unroll
      for (int j = 0; j < 4; ++j) {
        v16h b = load_b_g(&sW[0][(size_t)j * 16 * 128 + ks], 128, lane);
        cacc[j] = wmma_f16(a, b, cacc[j]);
      }
      if (DUAL) {
        v16h a2 = load_a_g(A2 + (size_t)m0 * K + kc + ks, K, lane);
#pragma unroll
        for (int j = 0; j < 4; ++j) {
          v16h b2 = load_b_g(&sW[1][(size_t)j * 16 * 128 + ks], 128, lane);
          cacc[j] = wmma_f16(a2, b2, cacc[j]);
        }
      }
    }
  }

  int rbase = m0 + ((lane >> 4) << 3);
#pragma unroll
  for (int j = 0; j < 4; ++j) {
    int   n  = n0 + j * 16 + (lane & 15);
    float bb = bias ? bias[n] : 0.0f;
    if (DUAL && bias2) bb += bias2[n];
    if (MODE == 2) {
      float* out = (float*)outv;
      for (int i = 0; i < 8; ++i)
        out[(size_t)(rbase + i) * N + n] = cacc[j][i] + bb;
    } else if (MODE == 1) {
      _Float16* out = (_Float16*)outv;   // [B, N, SEQ]
      for (int i = 0; i < 8; ++i) {
        int m  = rbase + i;
        int b_ = m >> 11, s = m & (SEQ - 1);
        out[((size_t)b_ * N + n) * SEQ + s] = (_Float16)(cacc[j][i] + bb);
      }
    } else {
      _Float16* out = (_Float16*)outv;
      for (int i = 0; i < 8; ++i)
        out[(size_t)(rbase + i) * N + n] = (_Float16)(cacc[j][i] + bb);
    }
  }
}

// ---------------------------------------------------------------- attention
// One block = (b, h, 16-query tile). 4 waves; each wave owns a 512-wide key strip
// for scores, then one 16-wide dh slice for ctx. Scores live in 64 KB LDS (f16).
__global__ __launch_bounds__(128) void attn_kernel(
    const _Float16* __restrict__ QT, const _Float16* __restrict__ Kp,
    const _Float16* __restrict__ Vt, const int* __restrict__ mask,
    float* __restrict__ attn_out, _Float16* __restrict__ ctx16) {
  __shared__ __align__(32) _Float16 sS[16 * SEQ];   // 65536 bytes

  int tid = threadIdx.x, lane = tid & 31, w = tid >> 5;
  int bid = blockIdx.x;
  int qt  = bid & 127;                  // SEQ/16 = 128 query tiles
  int h   = (bid >> 7) & (N_HEADS - 1);
  int b   = bid >> 11;

  // ---- phase 1: scores = (QT . K^T) / 8, masked, into LDS (f16)
  const _Float16* qbase = QT + ((size_t)(b * SEQ + qt * 16)) * D_MODEL + h * D_HEAD;
  v16h a0 = load_a_g(qbase, D_MODEL, lane);         // dh 0..31
  v16h a1 = load_a_g(qbase + 32, D_MODEL, lane);    // dh 32..63

  for (int kt = 0; kt < 32; ++kt) {
    int kcol = w * 512 + kt * 16;
    const _Float16* kb = Kp + ((size_t)(b * SEQ + kcol)) * D_MODEL + h * D_HEAD;
    v16h b0 = load_b_g(kb, D_MODEL, lane);
    v16h b1 = load_b_g(kb + 32, D_MODEL, lane);
    v8f c = {};
    c = wmma_f16(a0, b0, c);
    c = wmma_f16(a1, b1, c);

    int  coln = kcol + (lane & 15);
    bool msk  = (mask[(size_t)b * SEQ + coln] == 0);
    int  rb   = (lane >> 4) << 3;
    for (int i = 0; i < 8; ++i) {
      float v = msk ? -60000.0f : c[i] * 0.125f;    // 1/sqrt(64)
      sS[(rb + i) * SEQ + coln] = (_Float16)v;
    }
  }
  __syncthreads();

  // ---- softmax: 8 threads per row, wave-shuffle reductions (8-lane groups)
  int row = tid >> 3, sub = tid & 7;
  _Float16* rp = &sS[row * SEQ + sub * 256];

  float m = -1e30f;
  for (int j = 0; j < 256; ++j) m = fmaxf(m, (float)rp[j]);
  for (int d = 1; d < 8; d <<= 1) m = fmaxf(m, __shfl_xor(m, d, 32));

  float s = 0.0f;
  for (int j = 0; j < 256; ++j) {
    float e = __expf((float)rp[j] - m);
    rp[j] = (_Float16)e;
    s += e;
  }
  for (int d = 1; d < 8; d <<= 1) s += __shfl_xor(s, d, 32);
  float inv = 1.0f / s;

  {
    size_t abase = (((size_t)(b * N_HEADS + h) * SEQ) + (size_t)qt * 16 + row) * SEQ
                   + (size_t)sub * 256;
    for (int j = 0; j < 256; ++j) {
      float v = (float)rp[j] * inv;
      rp[j] = (_Float16)v;
      attn_out[abase + j] = v;
    }
  }
  __syncthreads();

  // ---- phase 2: ctx = attn(16 x 2048) . V(2048 x 64); wave w -> dh slice w*16
  {
    int dcol0 = h * D_HEAD + w * 16;
    v8f c = {};
    for (int ks = 0; ks < 64; ++ks) {
      int kb0 = ks * 32;
      int r = lane & 15, hi = (lane >> 4) << 3;
      const _Float16* p = &sS[r * SEQ + kb0 + hi];
      v8h lo = *(const v8h*)p;
      v8h hh = *(const v8h*)(p + 16);
      v16h a = pack16(lo, hh);
      const _Float16* vb = Vt + ((size_t)b * D_MODEL + dcol0) * SEQ + kb0;
      v16h bf = load_b_g(vb, SEQ, lane);
      c = wmma_f16(a, bf, c);
    }
    int rb = (lane >> 4) << 3;
    int dc = dcol0 + (lane & 15);
    for (int i = 0; i < 8; ++i) {
      int q = qt * 16 + rb + i;
      ctx16[((size_t)(b * SEQ + q)) * D_MODEL + dc] = (_Float16)c[i];
    }
  }
}

// ---------------------------------------------------------------- launch
extern "C" void kernel_launch(void* const* d_in, const int* in_sizes, int n_in,
                              void* d_out, int out_size, void* d_ws, size_t ws_size,
                              hipStream_t stream) {
  (void)in_sizes; (void)n_in; (void)out_size; (void)ws_size;
  const float* x    = (const float*)d_in[0];
  const float* ts   = (const float*)d_in[1];
  const int*   mask = (const int*)d_in[2];
  const float* tw   = (const float*)d_in[3];
  const float* tbv  = (const float*)d_in[4];
  const float* wq   = (const float*)d_in[5];
  const float* bq   = (const float*)d_in[6];
  const float* wk   = (const float*)d_in[7];
  const float* bk   = (const float*)d_in[8];
  const float* wv   = (const float*)d_in[9];
  const float* bv   = (const float*)d_in[10];
  const float* wt   = (const float*)d_in[11];
  const float* btb  = (const float*)d_in[12];
  const float* wo   = (const float*)d_in[13];
  const float* bo   = (const float*)d_in[14];

  char*  ws  = (char*)d_ws;
  size_t off = 0;
  auto alloc = [&](size_t bytes) -> char* {
    char* p = ws + off;
    off += (bytes + 255) & ~(size_t)255;
    return p;
  };
  const size_t actB = (size_t)M_TOT * D_MODEL * sizeof(_Float16);   // 8 MB
  const size_t wB   = (size_t)D_MODEL * D_MODEL * sizeof(_Float16); // 2 MB

  _Float16* x16   = (_Float16*)alloc(actB);
  _Float16* te16  = (_Float16*)alloc(actB);
  _Float16* xk16  = (_Float16*)alloc(actB);
  _Float16* QTb   = (_Float16*)alloc(actB);
  _Float16* Kpb   = (_Float16*)alloc(actB);
  _Float16* Vtb   = (_Float16*)alloc(actB);   // [B, D, SEQ] (transposed V)
  _Float16* ctx16 = (_Float16*)alloc(actB);
  _Float16* wq16  = (_Float16*)alloc(wB);
  _Float16* wk16  = (_Float16*)alloc(wB);
  _Float16* wv16  = (_Float16*)alloc(wB);
  _Float16* wt16  = (_Float16*)alloc(wB);
  _Float16* wo16  = (_Float16*)alloc(wB);

  float* outp  = (float*)d_out;                            // [B,S,D] fp32
  float* attnp = outp + (size_t)M_TOT * D_MODEL;           // [B,H,S,S] fp32

  const int nW = D_MODEL * D_MODEL;
  cvt_f32_to_f16<<<nW / 256, 256, 0, stream>>>(wq, wq16, nW);
  cvt_f32_to_f16<<<nW / 256, 256, 0, stream>>>(wk, wk16, nW);
  cvt_f32_to_f16<<<nW / 256, 256, 0, stream>>>(wv, wv16, nW);
  cvt_f32_to_f16<<<nW / 256, 256, 0, stream>>>(wt, wt16, nW);
  cvt_f32_to_f16<<<nW / 256, 256, 0, stream>>>(wo, wo16, nW);

  const int nAct = M_TOT * D_MODEL;
  prep_kernel<<<nAct / 256, 256, 0, stream>>>(x, ts, tw, tbv, x16, te16, xk16, nAct);

  const int gemmBlocks = (M_TOT / 128) * (D_MODEL / 64);   // 512
  gemm_wmma<1, 0><<<gemmBlocks, 256, 0, stream>>>(
      x16, wq16, te16, wt16, bq, btb, (void*)QTb, M_TOT, D_MODEL, D_MODEL);
  gemm_wmma<0, 0><<<gemmBlocks, 256, 0, stream>>>(
      xk16, wk16, nullptr, nullptr, bk, nullptr, (void*)Kpb, M_TOT, D_MODEL, D_MODEL);
  gemm_wmma<0, 1><<<gemmBlocks, 256, 0, stream>>>(
      x16, wv16, nullptr, nullptr, bv, nullptr, (void*)Vtb, M_TOT, D_MODEL, D_MODEL);

  attn_kernel<<<BATCH * N_HEADS * (SEQ / 16), 128, 0, stream>>>(
      QTb, Kpb, Vtb, mask, attnp, ctx16);

  gemm_wmma<0, 2><<<gemmBlocks, 256, 0, stream>>>(
      ctx16, wo16, nullptr, nullptr, bo, nullptr, (void*)outp, M_TOT, D_MODEL, D_MODEL);
}
